// MoELayer_12489764897382
// MI455X (gfx1250) — matrix operand order repaired and checked by reading, not compiled, not run on hardware
//
#include <hip/hip_runtime.h>
#include <hip/hip_bf16.h>
#include <stdint.h>

// ---------------------------------------------------------------------------
// MoE FFN (equal expert chunks) as two grouped GEMMs with bf16 WMMA, f32 acc.
//   H = gelu(X[e] @ W1[e] + b1[e])   -> bf16 in d_ws
//   Y = H @ W2[e] + b2[e]            -> fp32 in d_out
// E=8, T=1024 tokens/expert, D=1024, F=4096.
// Block tile 128x256, K-step 32, 256 threads = 8 wave32 waves (2x4), each
// wave owns a 64x64 tile = sixteen 16x16 f32 accumulators (4x4), so each
// ds_load_b128 feeds one WMMA. Ping-pong LDS buffers: staging of K-step
// kt+1 overlaps WMMA on kt, one barrier per step.
// ---------------------------------------------------------------------------

typedef __attribute__((ext_vector_type(16))) __bf16 v16bf;
typedef __attribute__((ext_vector_type(8)))  float  v8f;

#define NEXP   8
#define DMODEL 1024
#define DFF    4096
#define TPE    1024     // tokens per expert chunk
#define BM     128
#define BN     256
#define BK     32
#define LDSTR  40       // padded LDS row stride in ushorts (80B = 20 banks)
#define ABUF   (BM * LDSTR)   // one A ping-pong buffer, in ushorts
#define BBUF   (BN * LDSTR)   // one B ping-pong buffer, in ushorts

// Pack two fp32 into two bf16 (truncation) with one v_perm_b32.
__device__ __forceinline__ unsigned int pack_bf16x2(float lo, float hi) {
#if __has_builtin(__builtin_amdgcn_perm)
  // pool: bytes 0-3 = S1(lo), bytes 4-7 = S0(hi); take hi16 of each.
  return __builtin_amdgcn_perm(__float_as_uint(hi), __float_as_uint(lo),
                               0x07060302u);
#else
  unsigned int a = __float_as_uint(lo) >> 16;
  unsigned int b = __float_as_uint(hi) & 0xFFFF0000u;
  return a | b;
#endif
}

__device__ __forceinline__ unsigned short f32_to_bf16_rne(float f) {
  unsigned int u = __float_as_uint(f);
  unsigned int r = 0x7FFFu + ((u >> 16) & 1u);
  return (unsigned short)((u + r) >> 16);
}

__device__ __forceinline__ float gelu_tanh(float x) {
  const float k0 = 0.7978845608028654f;   // sqrt(2/pi)
  const float k1 = 0.044715f;
  float t = tanhf(k0 * (x + k1 * x * x * x));
  return 0.5f * x * (1.0f + t);
}

union FragU { v16bf v; uint4 q[2]; };

// A fragment, 16x32 bf16 (MxK): lanes 0-15 row=lane, K = [0..7]+[16..23];
// lanes 16-31 row=lane-16, K = [8..15]+[24..31].
__device__ __forceinline__ v16bf load_frag_a(const unsigned short* lds,
                                             int rowBase, int lane) {
  int row  = rowBase + (lane & 15);
  int koff = (lane >> 4) << 3;          // 0 or 8
  FragU f;
  f.q[0] = *(const uint4*)(lds + row * LDSTR + koff);
  f.q[1] = *(const uint4*)(lds + row * LDSTR + koff + 16);
  return f.v;
}

// B fragment, 32x16 bf16 (KxN), staged transposed in LDS as [n][k]:
// lanes 0-15 col=lane, K = [0..7]+[16..23]; lanes 16-31 K = [8..15]+[24..31].
__device__ __forceinline__ v16bf load_frag_b(const unsigned short* lds,
                                             int colBase, int lane) {
  int col  = colBase + (lane & 15);
  int koff = (lane >> 4) << 3;          // 0 or 8
  FragU f;
  f.q[0] = *(const uint4*)(lds + col * LDSTR + koff);
  f.q[1] = *(const uint4*)(lds + col * LDSTR + koff + 16);
  return f.v;
}

// Stage B tile 32(k) x 256(n) fp32 -> Bt[n][k] bf16. Each thread owns a
// 4-deep K column of one n: 4 coalesced b32 loads -> 2 v_perm -> one 8B
// contiguous LDS store.
__device__ __forceinline__ void stage_b(const float* __restrict__ wg,
                                        int ldw, int kk,
                                        unsigned short* bt, int t) {
#pragma unroll
  for (int i = 0; i < 8; ++i) {
    int idx = t + i * 256;              // 0..2047
    int n   = idx & 255;
    int kq  = (idx >> 8) << 2;          // 0,4,...,28
    const float* p = wg + (size_t)(kk + kq) * ldw + n;
    float f0 = p[0];
    float f1 = p[ldw];
    float f2 = p[2 * ldw];
    float f3 = p[3 * ldw];
    uint2 q;
    q.x = pack_bf16x2(f0, f1);
    q.y = pack_bf16x2(f2, f3);
    *(uint2*)(bt + n * LDSTR + kq) = q;
  }
}

// ---------------------------------------------------------------------------
// Kernel 1: H[e*T+m, n] = gelu(sum_k X[e*T+m,k] * W1[e,k,n] + b1[e,n])  (bf16)
// ---------------------------------------------------------------------------
__global__ __launch_bounds__(256) void moe_ffn1(
    const float* __restrict__ x,          // [E*T, D]
    const float* __restrict__ w1,         // [E, D, F]
    const float* __restrict__ b1,         // [E, F]
    unsigned short* __restrict__ h)       // [E*T, F] bf16
{
  __shared__ unsigned short As[2 * ABUF];
  __shared__ unsigned short Bt[2 * BBUF];

  const int nTilesN = DFF / BN;           // 16
  const int nTilesM = TPE / BM;           // 8
  int bid = blockIdx.x;
  int e   = bid / (nTilesM * nTilesN);
  int rem = bid % (nTilesM * nTilesN);
  int m0  = (rem / nTilesN) * BM;
  int n0  = (rem % nTilesN) * BN;

  int t     = threadIdx.x;
  int lane  = t & 31;
  int wave  = t >> 5;                     // 0..7
  int waveM = wave >> 2;                  // 0..1 -> 64-row strip
  int waveN = wave & 3;                   // 0..3 -> 64-col strip

  const float* xg = x  + (size_t)(e * TPE + m0) * DMODEL;
  const float* wg = w1 + (size_t)e * DMODEL * DFF + n0;

  v8f acc[4][4];
#pragma unroll
  for (int i = 0; i < 4; ++i)
#pragma unroll
    for (int j = 0; j < 4; ++j)
#pragma unroll
      for (int k = 0; k < 8; ++k) acc[i][j][k] = 0.0f;

  // Prologue: stage K-step 0 into buffer 0.
  {
#pragma unroll
    for (int i = 0; i < 4; ++i) {
      int idx = t + i * 256;
      int row = idx >> 3;
      int c4  = (idx & 7) << 2;
      float4 v = *(const float4*)(xg + (size_t)row * DMODEL + c4);
      uint2 q;
      q.x = pack_bf16x2(v.x, v.y);
      q.y = pack_bf16x2(v.z, v.w);
      *(uint2*)(As + row * LDSTR + c4) = q;
    }
    stage_b(wg, DFF, 0, Bt, t);
  }
  __syncthreads();

  const int NK = DMODEL / BK;             // 32
  for (int kt = 0; kt < NK; ++kt) {
    int curA = (kt & 1) * ABUF;
    int curB = (kt & 1) * BBUF;
    if (kt + 1 < NK) {
      int kk = (kt + 1) * BK;
      int nxtA = curA ^ ABUF;
      int nxtB = curB ^ BBUF;
#pragma unroll
      for (int i = 0; i < 4; ++i) {
        int idx = t + i * 256;
        int row = idx >> 3;
        int c4  = (idx & 7) << 2;
        float4 v = *(const float4*)(xg + (size_t)row * DMODEL + kk + c4);
        uint2 q;
        q.x = pack_bf16x2(v.x, v.y);
        q.y = pack_bf16x2(v.z, v.w);
        *(uint2*)(As + nxtA + row * LDSTR + c4) = q;
      }
      stage_b(wg, DFF, kk, Bt + nxtB, t);
    }

    v16bf af[4], bf[4];
#pragma unroll
    for (int tm = 0; tm < 4; ++tm)
      af[tm] = load_frag_a(As + curA, waveM * 64 + tm * 16, lane);
#pragma unroll
    for (int tn = 0; tn < 4; ++tn)
      bf[tn] = load_frag_b(Bt + curB, waveN * 64 + tn * 16, lane);
#pragma unroll
    for (int tm = 0; tm < 4; ++tm)
#pragma unroll
      for (int tn = 0; tn < 4; ++tn)
        acc[tm][tn] = __builtin_amdgcn_wmma_f32_16x16x32_bf16(
            false, af[tm], false, bf[tn], (short)0, acc[tm][tn], false, false);
    __syncthreads();
  }

  // Epilogue: bias + gelu, store bf16. Fully unrolled (no acc spills).
  const float* bias = b1 + (size_t)e * DFF + n0;
#pragma unroll
  for (int tm = 0; tm < 4; ++tm)
#pragma unroll
    for (int tn = 0; tn < 4; ++tn) {
      int col = waveN * 64 + tn * 16 + (lane & 15);
      float bv = bias[col];
#pragma unroll
      for (int j = 0; j < 8; ++j) {
        int row = waveM * 64 + tm * 16 + j + ((lane >> 4) << 3);
        float val = gelu_tanh(acc[tm][tn][j] + bv);
        h[(size_t)(e * TPE + m0 + row) * DFF + n0 + col] = f32_to_bf16_rne(val);
      }
    }
}

// ---------------------------------------------------------------------------
// Kernel 2: Y[e*T+m, n] = sum_k H[e*T+m,k] * W2[e,k,n] + b2[e,n]   (fp32 out)
// ---------------------------------------------------------------------------
__global__ __launch_bounds__(256) void moe_ffn2(
    const unsigned short* __restrict__ h, // [E*T, F] bf16
    const float* __restrict__ w2,         // [E, F, D]
    const float* __restrict__ b2,         // [E, D]
    float* __restrict__ y)                // [E*T, D]
{
  __shared__ unsigned short As[2 * ABUF];
  __shared__ unsigned short Bt[2 * BBUF];

  const int nTilesN = DMODEL / BN;        // 4
  const int nTilesM = TPE / BM;           // 8
  int bid = blockIdx.x;
  int e   = bid / (nTilesM * nTilesN);
  int rem = bid % (nTilesM * nTilesN);
  int m0  = (rem / nTilesN) * BM;
  int n0  = (rem % nTilesN) * BN;

  int t     = threadIdx.x;
  int lane  = t & 31;
  int wave  = t >> 5;
  int waveM = wave >> 2;                  // 0..1
  int waveN = wave & 3;                   // 0..3

  const unsigned short* hg = h + (size_t)(e * TPE + m0) * DFF;
  const float* wg = w2 + (size_t)e * DFF * DMODEL + n0;

  v8f acc[4][4];
#pragma unroll
  for (int i = 0; i < 4; ++i)
#pragma unroll
    for (int j = 0; j < 4; ++j)
#pragma unroll
      for (int k = 0; k < 8; ++k) acc[i][j][k] = 0.0f;

  // Prologue: stage K-step 0.
  {
#pragma unroll
    for (int i = 0; i < 2; ++i) {
      int idx = t + i * 256;
      int row = idx >> 2;
      int c8  = (idx & 3) << 3;
      uint4 v = *(const uint4*)(hg + (size_t)row * DFF + c8);
      *(uint4*)(As + row * LDSTR + c8) = v;
    }
    stage_b(wg, DMODEL, 0, Bt, t);
  }
  __syncthreads();

  const int NK = DFF / BK;                // 128
  for (int kt = 0; kt < NK; ++kt) {
    int curA = (kt & 1) * ABUF;
    int curB = (kt & 1) * BBUF;
    if (kt + 1 < NK) {
      int kk = (kt + 1) * BK;
      int nxtA = curA ^ ABUF;
      int nxtB = curB ^ BBUF;
#pragma unroll
      for (int i = 0; i < 2; ++i) {
        int idx = t + i * 256;
        int row = idx >> 2;
        int c8  = (idx & 3) << 3;
        uint4 v = *(const uint4*)(hg + (size_t)row * DFF + kk + c8);
        *(uint4*)(As + nxtA + row * LDSTR + c8) = v;
      }
      stage_b(wg, DMODEL, kk, Bt + nxtB, t);
    }

    v16bf af[4], bf[4];
#pragma unroll
    for (int tm = 0; tm < 4; ++tm)
      af[tm] = load_frag_a(As + curA, waveM * 64 + tm * 16, lane);
#pragma unroll
    for (int tn = 0; tn < 4; ++tn)
      bf[tn] = load_frag_b(Bt + curB, waveN * 64 + tn * 16, lane);
#pragma unroll
    for (int tm = 0; tm < 4; ++tm)
#pragma unroll
      for (int tn = 0; tn < 4; ++tn)
        acc[tm][tn] = __builtin_amdgcn_wmma_f32_16x16x32_bf16(
            false, af[tm], false, bf[tn], (short)0, acc[tm][tn], false, false);
    __syncthreads();
  }

  const float* bias = b2 + (size_t)e * DMODEL + n0;
#pragma unroll
  for (int tm = 0; tm < 4; ++tm)
#pragma unroll
    for (int tn = 0; tn < 4; ++tn) {
      int col = waveN * 64 + tn * 16 + (lane & 15);
      float bv = bias[col];
#pragma unroll
      for (int j = 0; j < 8; ++j) {
        int row = waveM * 64 + tm * 16 + j + ((lane >> 4) << 3);
        y[(size_t)(e * TPE + m0 + row) * DMODEL + n0 + col] =
            acc[tm][tn][j] + bv;
      }
    }
}

// ---------------------------------------------------------------------------
extern "C" void kernel_launch(void* const* d_in, const int* in_sizes, int n_in,
                              void* d_out, int out_size, void* d_ws, size_t ws_size,
                              hipStream_t stream) {
  const float* x  = (const float*)d_in[0];   // [2,4096,1024]
  const float* w1 = (const float*)d_in[1];   // [8,1024,4096]
  const float* b1 = (const float*)d_in[2];   // [8,4096]
  const float* w2 = (const float*)d_in[3];   // [8,4096,1024]
  const float* b2 = (const float*)d_in[4];   // [8,1024]
  float* out = (float*)d_out;                // [2,4096,1024]

  // Intermediate H: 8192 x 4096 bf16 = 64 MiB in workspace.
  unsigned short* hbuf = (unsigned short*)d_ws;

  dim3 blk(256);
  int grid1 = NEXP * (TPE / BM) * (DFF / BN);     // 8*8*16 = 1024 blocks
  int grid2 = NEXP * (TPE / BM) * (DMODEL / BN);  // 8*8*4  = 256 blocks

  moe_ffn1<<<grid1, blk, 0, stream>>>(x, w1, b1, hbuf);
  moe_ffn2<<<grid2, blk, 0, stream>>>(hbuf, w2, b2, out);
}
